// RelevancyWeightedSum1D_26268019982849
// MI455X (gfx1250) — compile-verified, matrix operand order checked by reading, not run on hardware
//
#include <hip/hip_runtime.h>
#include <hip/hip_bf16.h>

// ---------------------------------------------------------------------------
// Problem constants (match the reference)
// ---------------------------------------------------------------------------
#define B_    4
#define L_    2048
#define C_    512
#define H_    8
#define CH_   64          // C / H
#define ROWS_ (B_ * L_)   // 8192
#define NQKV_ (3 * C_)    // 1536
#define SCALE_ 0.125f     // 1/sqrt(CH)

typedef __attribute__((ext_vector_type(16))) __bf16 v16bf;
typedef __attribute__((ext_vector_type(8)))  __bf16 v8bf;
typedef __attribute__((ext_vector_type(8)))  float  v8f;
typedef __attribute__((ext_vector_type(4)))  float  v4f;

// float -> bf16 round-to-nearest-even
static __device__ __forceinline__ __bf16 f2bf(float f) {
  union { float f; unsigned u; } cv; cv.f = f;
  unsigned r = cv.u + 0x7FFFu + ((cv.u >> 16) & 1u);
  unsigned short hs = (unsigned short)(r >> 16);
  __bf16 o;
  __builtin_memcpy(&o, &hs, sizeof(o));
  return o;
}

// ---------------------------------------------------------------------------
// 16x32 bf16 fragment in the CDNA5 16-bit A-matrix VGPR layout.
//   lane = M (0..15) + 16g ; K runs {k0+8g..+7, k0+16+8g..+7}.
// Also used for B fragments from N-major (transposed) sources.
// ---------------------------------------------------------------------------
static __device__ __forceinline__ v16bf load_frag(const __bf16* base,
                                                  int stride, int row, int k0, int lane) {
  const int m = lane & 15, g = lane >> 4;
  const __bf16* p = base + (size_t)(row + m) * stride + k0 + 8 * g;
  v8bf lo = *(const v8bf*)(p);
  v8bf hi = *(const v8bf*)(p + 16);
  v16bf f;
#pragma unroll
  for (int i = 0; i < 8; ++i) { f[i] = lo[i]; f[i + 8] = hi[i]; }
  return f;
}

static __device__ __forceinline__ v8f wmma_bf16(v16bf a, v16bf b, v8f c) {
  return __builtin_amdgcn_wmma_f32_16x16x32_bf16(false, a, false, b, (short)0, c,
                                                 false, false);
}

// ---------------------------------------------------------------------------
// DPP lane swaps (VALU only). Butterfly over 16 lanes.
// ---------------------------------------------------------------------------
template <int CTRL>
static __device__ __forceinline__ float dpp_mov(float x) {
  int r = __builtin_amdgcn_update_dpp(0, __builtin_bit_cast(int, x),
                                      CTRL, 0xF, 0xF, true);
  return __builtin_bit_cast(float, r);
}
static __device__ __forceinline__ float red16_max(float m) {
  m = fmaxf(m, dpp_mov<0xB1>(m));   // xor 1
  m = fmaxf(m, dpp_mov<0x4E>(m));   // xor 2
  m = fmaxf(m, dpp_mov<0x141>(m));  // row_half_mirror -> xor 4 step
  m = fmaxf(m, dpp_mov<0x140>(m));  // row_mirror      -> xor 8 step
  return m;
}
static __device__ __forceinline__ float red16_sum(float s) {
  s += dpp_mov<0xB1>(s);
  s += dpp_mov<0x4E>(s);
  s += dpp_mov<0x141>(s);
  s += dpp_mov<0x140>(s);
  return s;
}

// ---------------------------------------------------------------------------
// CDNA5 async global->LDS copy (ASYNCcnt path), 16 bytes per call.
// ---------------------------------------------------------------------------
static __device__ __forceinline__ unsigned lds_off(const void* p) {
  return (unsigned)(unsigned long long)p;   // low 32 bits of shared aperture
}
static __device__ __forceinline__ void async_copy16(unsigned lds_byte_off,
                                                    const void* gptr) {
  asm volatile("global_load_async_to_lds_b128 %0, %1, off"
               :: "v"(lds_byte_off), "v"(gptr) : "memory");
}
// wait until <=4 outstanding: with 4 chunks/thread/tile and in-order async
// completion, this means "previous tile has fully landed".
static __device__ __forceinline__ void wait_async4() {
  asm volatile("s_wait_asynccnt 0x4" ::: "memory");
}

// ---------------------------------------------------------------------------
// Kernel 0: f32 (K x N) weight -> bf16 transposed (N x K). 2-D grid, no div.
// ---------------------------------------------------------------------------
__global__ void __launch_bounds__(256) cvt_transpose_kernel(
    const float* __restrict__ w, __bf16* __restrict__ wT, int K, int N) {
  const int n = blockIdx.x * 256 + threadIdx.x;
  const int k = blockIdx.y;
  wT[(size_t)n * K + k] = f2bf(w[(size_t)k * N + n]);
}

// ---------------------------------------------------------------------------
// Kernel 1: LayerNorm, one wave32 per 512-float row, bf16 output
// ---------------------------------------------------------------------------
__global__ void __launch_bounds__(256) ln_kernel(
    const float* __restrict__ x, const float* __restrict__ gamma,
    const float* __restrict__ beta, __bf16* __restrict__ xn) {
  const int lane = threadIdx.x & 31;
  const int wave = threadIdx.x >> 5;
  const int row  = blockIdx.x * 8 + wave;
  const float* xr = x + (size_t)row * C_;

  float v[16];
  float s = 0.f, ss = 0.f;
#pragma unroll
  for (int i = 0; i < 4; ++i) {
    v4f t = *(const v4f*)(xr + i * 128 + lane * 4);
#pragma unroll
    for (int c = 0; c < 4; ++c) {
      float f = t[c];
      v[i * 4 + c] = f;
      s += f; ss += f * f;
    }
  }
#pragma unroll
  for (int off = 16; off >= 1; off >>= 1) {
    s  += __shfl_xor(s,  off, 32);
    ss += __shfl_xor(ss, off, 32);
  }
  const float mu   = s * (1.0f / C_);
  const float var  = ss * (1.0f / C_) - mu * mu;
  const float rinv = rsqrtf(var + 1e-5f);

  __bf16* outr = xn + (size_t)row * C_;
#pragma unroll
  for (int i = 0; i < 4; ++i) {
    const int j0 = i * 128 + lane * 4;
#pragma unroll
    for (int c = 0; c < 4; ++c) {
      const int j = j0 + c;
      outr[j] = f2bf((v[i * 4 + c] - mu) * rinv * gamma[j] + beta[j]);
    }
  }
}

// ---------------------------------------------------------------------------
// Kernel 2: bf16 WMMA GEMM (M x K) @ (K x N), B pre-transposed (N x K).
// 256 threads = 8 waves, 128x128 block tile, 32x64 per wave (8 accumulators).
// A/B k-slabs (128x32 each) double-buffered in LDS via async copies; the
// next slab streams in while WMMAs consume the current one.
// MODE 0: QKV scatter (q scaled, k [BH][L][CH], v transposed [BH][CH][L])
// MODE 1: f32 out = acc + bias + residual
// ---------------------------------------------------------------------------
template <int MODE>
__global__ void __launch_bounds__(256) gemm_wmma_kernel(
    const __bf16* __restrict__ A, const __bf16* __restrict__ Bt,
    const float* __restrict__ bias, const float* __restrict__ resid,
    float* __restrict__ outF,
    __bf16* __restrict__ outQ, __bf16* __restrict__ outK, __bf16* __restrict__ outV,
    int Kdim) {
  const int tid  = threadIdx.x;
  const int lane = tid & 31;
  const int wave = tid >> 5;
  const int m0 = blockIdx.x * 128 + (wave >> 1) * 32;   // global row base of wave
  const int n0 = blockIdx.y * 128 + (wave & 1) * 64;    // global col base of wave
  const int mrel = (wave >> 1) * 32;                    // within block tile
  const int nrel = (wave & 1) * 64;

  __shared__ __bf16 ldsA[2 * 128 * 32];   // 2 x 8KB
  __shared__ __bf16 ldsB[2 * 128 * 32];   // 2 x 8KB
  const unsigned offA0 = lds_off(ldsA);
  const unsigned offB0 = lds_off(ldsB);
  const __bf16* Ab = A  + (size_t)(blockIdx.x * 128) * Kdim;
  const __bf16* Bb = Bt + (size_t)(blockIdx.y * 128) * Kdim;

  // stage one 128x32 k-slab of A and B (4 x 16B chunks per thread)
  auto stage = [&](unsigned buf, int k0) {
#pragma unroll
    for (int j = 0; j < 2; ++j) {
      const unsigned cc = tid + 256 * j;          // 512 chunks per matrix
      const unsigned r = cc >> 2, c = cc & 3;
      const unsigned lo = buf * 8192 + (r * 32 + c * 8) * 2;
      async_copy16(offA0 + lo, Ab + (size_t)r * Kdim + k0 + c * 8);
      async_copy16(offB0 + lo, Bb + (size_t)r * Kdim + k0 + c * 8);
    }
  };

  const int nk = Kdim / 32;
  stage(0, 0);

  v8f acc[2][4] = {};
  for (int ki = 0; ki < nk; ++ki) {
    const int knext = (ki + 1 < nk) ? (ki + 1) * 32 : ki * 32;  // clamped prefetch
    stage((ki + 1) & 1, knext);
    wait_async4();                    // current slab has landed
    __syncthreads();

    const __bf16* cA = ldsA + (ki & 1) * 4096;
    const __bf16* cB = ldsB + (ki & 1) * 4096;
    v16bf a0 = load_frag(cA, 32, mrel,      0, lane);
    v16bf a1 = load_frag(cA, 32, mrel + 16, 0, lane);
    v16bf b0 = load_frag(cB, 32, nrel,      0, lane);
    v16bf b1 = load_frag(cB, 32, nrel + 16, 0, lane);
    v16bf b2 = load_frag(cB, 32, nrel + 32, 0, lane);
    v16bf b3 = load_frag(cB, 32, nrel + 48, 0, lane);
    acc[0][0] = wmma_bf16(a0, b0, acc[0][0]);
    acc[0][1] = wmma_bf16(a0, b1, acc[0][1]);
    acc[0][2] = wmma_bf16(a0, b2, acc[0][2]);
    acc[0][3] = wmma_bf16(a0, b3, acc[0][3]);
    acc[1][0] = wmma_bf16(a1, b0, acc[1][0]);
    acc[1][1] = wmma_bf16(a1, b1, acc[1][1]);
    acc[1][2] = wmma_bf16(a1, b2, acc[1][2]);
    acc[1][3] = wmma_bf16(a1, b3, acc[1][3]);
    __syncthreads();                  // everyone done reading before overwrite
  }

  // ---- epilogue: b / three / h are wave-uniform; hoist them ---------------
  const int g = lane >> 4, col = lane & 15;
  if (MODE == 0) {
    const int bq = m0 >> 11;                 // batch (32-row span never crosses)
    const int l0 = m0 & (L_ - 1);
    const int three = n0 >> 9;               // q/k/v (64-col span never crosses)
    const int h  = (n0 >> 6) & 7;
    const int bh = bq * H_ + h;
    if (three == 0) {
      __bf16* dst = outQ + ((size_t)bh * L_ + l0) * CH_;
#pragma unroll
      for (int rr = 0; rr < 2; ++rr)
#pragma unroll
        for (int t = 0; t < 4; ++t)
#pragma unroll
          for (int r = 0; r < 8; ++r)
            dst[(16 * rr + r + 8 * g) * CH_ + 16 * t + col] =
                f2bf((acc[rr][t][r] + bias[n0 + 16 * t + col]) * SCALE_);
    } else if (three == 1) {
      __bf16* dst = outK + ((size_t)bh * L_ + l0) * CH_;
#pragma unroll
      for (int rr = 0; rr < 2; ++rr)
#pragma unroll
        for (int t = 0; t < 4; ++t)
#pragma unroll
          for (int r = 0; r < 8; ++r)
            dst[(16 * rr + r + 8 * g) * CH_ + 16 * t + col] =
                f2bf(acc[rr][t][r] + bias[n0 + 16 * t + col]);
    } else {
      __bf16* dst = outV + (size_t)bh * CH_ * L_ + l0;   // transposed V
#pragma unroll
      for (int rr = 0; rr < 2; ++rr)
#pragma unroll
        for (int t = 0; t < 4; ++t)
#pragma unroll
          for (int r = 0; r < 8; ++r)
            dst[(size_t)(16 * t + col) * L_ + 16 * rr + r + 8 * g] =
                f2bf(acc[rr][t][r] + bias[n0 + 16 * t + col]);
    }
  } else {
    float* po = outF + (size_t)m0 * C_ + n0;
    const float* pi = resid + (size_t)m0 * C_ + n0;
#pragma unroll
    for (int rr = 0; rr < 2; ++rr)
#pragma unroll
      for (int t = 0; t < 4; ++t)
#pragma unroll
        for (int r = 0; r < 8; ++r) {
          const int o = (16 * rr + r + 8 * g) * C_ + 16 * t + col;
          po[o] = acc[rr][t][r] + bias[n0 + 16 * t + col] + pi[o];
        }
  }
}

// ---------------------------------------------------------------------------
// Kernel 3: flash attention. Block = (b,h) x 128 q-rows, 8 waves x 16 rows.
// kv tiles of 64, K/V double-buffered in LDS via async copies (prefetch tile
// i+1 while computing tile i), 16 WMMAs per tile, DPP online softmax.
// ---------------------------------------------------------------------------
__global__ void __launch_bounds__(256) attn_kernel(
    const __bf16* __restrict__ Q, const __bf16* __restrict__ Kmat,
    const __bf16* __restrict__ Vt, __bf16* __restrict__ Z) {
  const int bh = blockIdx.x;                 // 0..31
  const int b = bh >> 3, h = bh & 7;
  const int tid = threadIdx.x;
  const int lane = tid & 31;
  const int wave = tid >> 5;
  const int q0 = blockIdx.y * 128 + wave * 16;

  const __bf16* Qp = Q    + (size_t)bh * L_ * CH_;   // [L][CH]
  const __bf16* Kp = Kmat + (size_t)bh * L_ * CH_;   // [L][CH]
  const __bf16* Vp = Vt   + (size_t)bh * CH_ * L_;   // [CH][L]

  __shared__ __bf16 ldsK[2 * 64 * 64];       // 2 x 8KB, kv rows (contiguous)
  __shared__ __bf16 ldsV[2 * 64 * 64];       // 2 x 8KB, ch rows x kv cols
  __shared__ __bf16 ldsP[8][16 * 64];        // per-wave P staging, 16KB
  __bf16* myP = &ldsP[wave][0];
  const unsigned offK0 = lds_off(ldsK);
  const unsigned offV0 = lds_off(ldsV);

  // stage one kv-64 tile of K and V (4 x 16B chunks per thread)
  auto stage = [&](unsigned buf, int kv) {
#pragma unroll
    for (int j = 0; j < 2; ++j) {
      const unsigned c = tid + 256 * j;              // 512 chunks per matrix
      async_copy16(offK0 + buf * 8192 + c * 16, Kp + (size_t)kv * CH_ + c * 8);
      const unsigned vr = c >> 3, vc = c & 7;
      async_copy16(offV0 + buf * 8192 + (vr * 64 + vc * 8) * 2,
                   Vp + (size_t)vr * L_ + kv + vc * 8);
    }
  };

  v16bf qf0 = load_frag(Qp, CH_, q0, 0,  lane);
  v16bf qf1 = load_frag(Qp, CH_, q0, 32, lane);

  v8f o0 = {}, o1 = {}, o2 = {}, o3 = {};
  float rowmax[8], rowsum[8];
#pragma unroll
  for (int r = 0; r < 8; ++r) { rowmax[r] = -3.0e38f; rowsum[r] = 0.f; }

  const int g = lane >> 4, cl = lane & 15;

  stage(0, 0);
  for (int kv = 0, it = 0; kv < L_; kv += 64, ++it) {
    const int knext = (kv + 64 < L_) ? kv + 64 : kv;   // clamped prefetch
    stage((it + 1) & 1, knext);
    wait_async4();                  // tile `it` has landed
    __syncthreads();

    const __bf16* cK = ldsK + (it & 1) * 4096;
    const __bf16* cV = ldsV + (it & 1) * 4096;

    // ---- S = Q . K^T (16 x 64), 8 WMMAs -----------------------------------
    v8f s0 = {}, s1 = {}, s2 = {}, s3 = {};
#pragma unroll
    for (int s = 0; s < 2; ++s) {
      v16bf qf = s ? qf1 : qf0;
      s0 = wmma_bf16(qf, load_frag(cK, 64, 0,  32 * s, lane), s0);
      s1 = wmma_bf16(qf, load_frag(cK, 64, 16, 32 * s, lane), s1);
      s2 = wmma_bf16(qf, load_frag(cK, 64, 32, 32 * s, lane), s2);
      s3 = wmma_bf16(qf, load_frag(cK, 64, 48, 32 * s, lane), s3);
    }

    // ---- online softmax over 64 cols (DPP butterflies) --------------------
#pragma unroll
    for (int r = 0; r < 8; ++r) {
      float m = fmaxf(fmaxf(s0[r], s1[r]), fmaxf(s2[r], s3[r]));
      m = red16_max(m);
      const float mnew = fmaxf(rowmax[r], m);
      const float corr = __expf(rowmax[r] - mnew);
      rowmax[r] = mnew;
      const float p0 = __expf(s0[r] - mnew);
      const float p1 = __expf(s1[r] - mnew);
      const float p2 = __expf(s2[r] - mnew);
      const float p3 = __expf(s3[r] - mnew);
      __bf16* pr = myP + (r + 8 * g) * 64 + cl;
      pr[0]  = f2bf(p0);
      pr[16] = f2bf(p1);
      pr[32] = f2bf(p2);
      pr[48] = f2bf(p3);
      const float ps = red16_sum((p0 + p1) + (p2 + p3));
      rowsum[r] = rowsum[r] * corr + ps;
      o0[r] *= corr; o1[r] *= corr; o2[r] *= corr; o3[r] *= corr;
    }

    // ---- O += P . V (K = 64 kv, N = 64 ch), 8 WMMAs -----------------------
    v16bf pf0 = load_frag(myP, 64, 0, 0,  lane);
    v16bf pf1 = load_frag(myP, 64, 0, 32, lane);
#pragma unroll
    for (int s = 0; s < 2; ++s) {
      v16bf pf = s ? pf1 : pf0;
      o0 = wmma_bf16(pf, load_frag(cV, 64, 0,  32 * s, lane), o0);
      o1 = wmma_bf16(pf, load_frag(cV, 64, 16, 32 * s, lane), o1);
      o2 = wmma_bf16(pf, load_frag(cV, 64, 32, 32 * s, lane), o2);
      o3 = wmma_bf16(pf, load_frag(cV, 64, 48, 32 * s, lane), o3);
    }
    __syncthreads();   // all waves done with this buffer before next overwrite
  }

  // ---- normalize and write z as bf16 [B][L][C] ----------------------------
#pragma unroll
  for (int r = 0; r < 8; ++r) {
    const float inv = 1.0f / rowsum[r];
    const int row = q0 + r + 8 * g;
    __bf16* zp = Z + ((size_t)b * L_ + row) * C_ + h * CH_;
    zp[cl]      = f2bf(o0[r] * inv);
    zp[16 + cl] = f2bf(o1[r] * inv);
    zp[32 + cl] = f2bf(o2[r] * inv);
    zp[48 + cl] = f2bf(o3[r] * inv);
  }
}

// ---------------------------------------------------------------------------
// Launch: cvt -> LN -> QKV GEMM -> flash attention -> out GEMM (+residual)
// ---------------------------------------------------------------------------
extern "C" void kernel_launch(void* const* d_in, const int* in_sizes, int n_in,
                              void* d_out, int out_size, void* d_ws, size_t ws_size,
                              hipStream_t stream) {
  (void)in_sizes; (void)n_in; (void)out_size; (void)ws_size;
  const float* input_x = (const float*)d_in[0];
  const float* gamma   = (const float*)d_in[1];
  const float* beta    = (const float*)d_in[2];
  const float* w_xyz   = (const float*)d_in[3];
  const float* b_xyz   = (const float*)d_in[4];
  const float* w_end   = (const float*)d_in[5];
  const float* b_end   = (const float*)d_in[6];
  float* out = (float*)d_out;

  __bf16* ws = (__bf16*)d_ws;
  size_t off = 0;
  __bf16* xn  = ws + off; off += (size_t)ROWS_ * C_;
  __bf16* wTx = ws + off; off += (size_t)NQKV_ * C_;
  __bf16* wTe = ws + off; off += (size_t)C_ * C_;
  __bf16* q   = ws + off; off += (size_t)B_ * H_ * L_ * CH_;
  __bf16* k   = ws + off; off += (size_t)B_ * H_ * L_ * CH_;
  __bf16* vT  = ws + off; off += (size_t)B_ * H_ * L_ * CH_;
  __bf16* z   = ws + off; off += (size_t)ROWS_ * C_;

  cvt_transpose_kernel<<<dim3(NQKV_ / 256, C_), 256, 0, stream>>>(w_xyz, wTx, C_, NQKV_);
  cvt_transpose_kernel<<<dim3(C_ / 256, C_),    256, 0, stream>>>(w_end, wTe, C_, C_);

  ln_kernel<<<ROWS_ / 8, 256, 0, stream>>>(input_x, gamma, beta, xn);

  gemm_wmma_kernel<0><<<dim3(ROWS_ / 128, NQKV_ / 128), 256, 0, stream>>>(
      xn, wTx, b_xyz, nullptr, nullptr, q, k, vT, C_);

  attn_kernel<<<dim3(B_ * H_, L_ / 128), 256, 0, stream>>>(q, k, vT, z);

  gemm_wmma_kernel<1><<<dim3(ROWS_ / 128, C_ / 128), 256, 0, stream>>>(
      z, wTe, b_end, input_x, out, nullptr, nullptr, nullptr, C_);
}
